// QuantumOperator_55473797595914
// MI455X (gfx1250) — matrix operand order repaired and checked by reading, not compiled, not run on hardware
//
#include <hip/hip_runtime.h>

typedef _Float16 half_t;
typedef __attribute__((ext_vector_type(16))) _Float16 v16h;
typedef __attribute__((ext_vector_type(8)))  _Float16 v8h;
typedef __attribute__((ext_vector_type(8)))  float    v8f;

#define QDIM     128
#define LDB      136                   // padded LDS row stride (halves): breaks 64-bank pattern
#define M_TOTAL  (16 * 16384)          // B * S rows
#define SLABS_PER_BLOCK 8              // 8 waves * 16 rows = 128 rows per workgroup

// Load 16 f16 A-fragment elements for this lane: 8 floats at base, 8 floats at base+16,
// matching the CDNA5 16-bit A-matrix 16x32 layout (lane<16: K{0..7,16..23}, lane>=16: K{8..15,24..31}
// after the caller applies the (lane>>4)*8 column offset).
__device__ __forceinline__ v16h load_a_f16(const float* __restrict__ base) {
    const float4* q0 = (const float4*)(base);
    const float4* q1 = (const float4*)(base + 16);
    float4 f0 = q0[0], f1 = q0[1];
    float4 f2 = q1[0], f3 = q1[1];
    v16h a;
    a[0]  = (half_t)f0.x; a[1]  = (half_t)f0.y; a[2]  = (half_t)f0.z; a[3]  = (half_t)f0.w;
    a[4]  = (half_t)f1.x; a[5]  = (half_t)f1.y; a[6]  = (half_t)f1.z; a[7]  = (half_t)f1.w;
    a[8]  = (half_t)f2.x; a[9]  = (half_t)f2.y; a[10] = (half_t)f2.z; a[11] = (half_t)f2.w;
    a[12] = (half_t)f3.x; a[13] = (half_t)f3.y; a[14] = (half_t)f3.z; a[15] = (half_t)f3.w;
    return a;
}

// Load 8 f16 at base and 8 f16 at base+16 from LDS (16B-aligned) as a B fragment.
__device__ __forceinline__ v16h load_b_lds(const half_t* base) {
    const v8h* q = (const v8h*)base;
    v8h b0 = q[0], b1 = q[2];
    v16h b;
#pragma unroll
    for (int i = 0; i < 8; ++i) { b[i] = b0[i]; b[i + 8] = b1[i]; }
    return b;
}

__global__ __launch_bounds__(256) void qop_wmma_f16(const float* __restrict__ sr,
                                                    const float* __restrict__ si,
                                                    const float* __restrict__ prm,
                                                    float* __restrict__ out) {
    // Operator images in LDS, f16, stored transposed (row index = N, col index = K):
    //   sOr[n*LDB + k] = Or[k][n]  (Or symmetric, so == Or[n][k])
    //   sOi[n*LDB + k] = Oi[k][n]  = p_k - p_n
    __shared__ half_t sOr[QDIM * LDB];
    __shared__ half_t sOi[QDIM * LDB];
    __shared__ float  sp[QDIM];

    const int tid = threadIdx.x;
    if (tid < QDIM) sp[tid] = prm[tid];
    __syncthreads();

#pragma unroll 4
    for (int idx = tid; idx < QDIM * QDIM; idx += 256) {
        const int n = idx >> 7;
        const int k = idx & 127;
        const float pn = sp[n], pk = sp[k];
        const float orv = (n == k) ? (1.0f / (1.0f + __expf(-pn))) : (pn * pk);
        sOr[n * LDB + k] = (half_t)orv;
        sOi[n * LDB + k] = (half_t)(pk - pn);
    }
    __syncthreads();

    const int lane = tid & 31;
    const int wave = tid >> 5;
    const int l15  = lane & 15;
    const int hi   = lane >> 4;                         // 0: lanes 0-15, 1: lanes 16-31

    const int  slab = blockIdx.x * SLABS_PER_BLOCK + wave;  // 16-row slab
    const int  row0 = slab * 16;
    const long rowA = (long)(row0 + l15) * QDIM;            // this lane's A-row base

    v8f acc_re[8], acc_im[8];
    const v8f vzero = {};
#pragma unroll
    for (int n = 0; n < 8; ++n) { acc_re[n] = vzero; acc_im[n] = vzero; }

#pragma unroll
    for (int k = 0; k < 4; ++k) {                       // K chunks of 32
        const int c0 = k * 32 + hi * 8;                 // A column base for this lane
        const v16h a_sr  = load_a_f16(sr + rowA + c0);
        const v16h a_si  = load_a_f16(si + rowA + c0);
        const v16h a_nsi = -a_si;                       // for out_real = sr@Or + (-si)@Oi
        const int bcol = k * 32 + hi * 8;               // B K-offset for this lane half
#pragma unroll
        for (int n = 0; n < 8; ++n) {                   // N tiles of 16
            const int  boff = (n * 16 + l15) * LDB + bcol;
            const v16h b_or = load_b_lds(&sOr[boff]);
            const v16h b_oi = load_b_lds(&sOi[boff]);
            acc_re[n] = __builtin_amdgcn_wmma_f32_16x16x32_f16(
                false, a_sr,  false, b_or, (short)0, acc_re[n], false, false);
            acc_re[n] = __builtin_amdgcn_wmma_f32_16x16x32_f16(
                false, a_nsi, false, b_oi, (short)0, acc_re[n], false, false);
            acc_im[n] = __builtin_amdgcn_wmma_f32_16x16x32_f16(
                false, a_sr,  false, b_oi, (short)0, acc_im[n], false, false);
            acc_im[n] = __builtin_amdgcn_wmma_f32_16x16x32_f16(
                false, a_si,  false, b_or, (short)0, acc_im[n], false, false);
        }
    }

    // D layout: element v -> row (v + hi*8), col = l15 within the 16x16 tile.
    float* __restrict__ out_re = out;
    float* __restrict__ out_im = out + (long)M_TOTAL * QDIM;
#pragma unroll
    for (int v = 0; v < 8; ++v) {
        const long r = (long)(row0 + v + hi * 8) * QDIM + l15;
#pragma unroll
        for (int n = 0; n < 8; ++n) {
            out_re[r + n * 16] = acc_re[n][v];
            out_im[r + n * 16] = acc_im[n][v];
        }
    }
}

extern "C" void kernel_launch(void* const* d_in, const int* in_sizes, int n_in,
                              void* d_out, int out_size, void* d_ws, size_t ws_size,
                              hipStream_t stream) {
    const float* sr  = (const float*)d_in[0];   // state_real [16,16384,128] f32
    const float* si  = (const float*)d_in[1];   // state_imag [16,16384,128] f32
    const float* prm = (const float*)d_in[2];   // params [128] f32
    float* out = (float*)d_out;                 // out_real ++ out_imag, f32

    const int n_blocks = M_TOTAL / (16 * SLABS_PER_BLOCK);  // 2048
    qop_wmma_f16<<<n_blocks, 256, 0, stream>>>(sr, si, prm, out);
}
